// GreekTextCollapseModel_54262616818142
// MI455X (gfx1250) — matrix operand order repaired and checked by reading, not compiled, not run on hardware
//
#include <hip/hip_runtime.h>
#include <hip/hip_bf16.h>

// ---------------------------------------------------------------------------
// Seq2seq LSTM w/ attention on MI455X (gfx1250, wave32, WMMA bf16->f32).
// V=48 H=256 L=3 B=128 S=T=512.
//  * Input-side LSTM GEMMs hoisted out of the time scans (M=65536 WMMA GEMMs)
//  * Attention softmax is invariant to the decoder state (hq term is constant
//    over s), so context / ctx-gates / ctx-output are computed once.
//  * Recurrent step = fused WMMA GEMM (128x1024, K=256[+256]) + gate epilogue.
//  * GEMM K-loop unrolled by 2 with alternating fragment buffers: loads for
//    the next k-tile overlap current-tile WMMAs with no register-rotation
//    copies (avoids the v_dual_mov flood between v_wmma ops).
// ---------------------------------------------------------------------------

#define VV 48
#define HH 256
#define BB 128
#define SS 512
#define TT 512
#define G4H 1024   // 4*H

typedef __attribute__((ext_vector_type(16))) __bf16 v16bf;
typedef __attribute__((ext_vector_type(8)))  __bf16 v8bf;
typedef __attribute__((ext_vector_type(8)))  float  v8f;

// ---- WMMA fragment loaders (CDNA5 ISA 7.12.2 layouts, wave32) --------------

// A: 16x32 bf16, row-major, row = lane&15.
// lanes 0-15 hold K {0..7,16..23}; lanes 16-31 hold K {8..15,24..31}.
__device__ __forceinline__ v16bf load_a_frag(const __bf16* __restrict__ A,
                                             int lda, int row0, int k0, int lane) {
  int m  = lane & 15;
  int ks = (lane >> 4) << 3;                 // 0 or 8
  const __bf16* base = A + (size_t)(row0 + m) * lda + k0 + ks;
  v8bf lo = *(const v8bf*)(base);
  v8bf hi = *(const v8bf*)(base + 16);
  v16bf r;
#pragma unroll
  for (int i = 0; i < 8; ++i) { r[i] = lo[i]; r[i + 8] = hi[i]; }
  return r;
}

// B: 32x16 bf16 (KxN) sourced from W stored [N][ldw] row-major (so W^T):
// lane&15 = column n; lanes 0-15 hold K {0..7,16..23}; lanes 16-31 hold K {8..15,24..31}.
__device__ __forceinline__ v16bf load_b_frag(const __bf16* __restrict__ W,
                                             int ldw, int n0, int k0, int lane) {
  int n  = lane & 15;
  int kb = (lane >> 4) << 3;                 // 0 or 8
  const __bf16* base = W + (size_t)(n0 + n) * ldw + k0 + kb;
  v8bf lo = *(const v8bf*)(base);
  v8bf hi = *(const v8bf*)(base + 16);
  v16bf r;
#pragma unroll
  for (int i = 0; i < 8; ++i) { r[i] = lo[i]; r[i + 8] = hi[i]; }
  return r;
}

__device__ __forceinline__ v8f wmma_bf16(v16bf a, v16bf b, v8f c) {
  return __builtin_amdgcn_wmma_f32_16x16x32_bf16(false, a, false, b, (short)0, c,
                                                 false, false);
}

__device__ __forceinline__ float sigf(float x) { return 1.0f / (1.0f + __expf(-x)); }

// ---- Big hoisted GEMM: C[M,N] = A[M,K](bf16) @ W[N,K]^T(bf16) + bias -------
// block = 256 thr (8 waves); tile = 64M x 128N; wave = 16M x 64N (4 accums).
// K-loop unrolled by 2 with alternating fragment buffers (K % 64 == 0):
// loads refill one buffer while WMMAs consume the other -> no rotation movs.
__global__ __launch_bounds__(256) void gemm_bias_kernel(
    const __bf16* __restrict__ A, int lda,
    const __bf16* __restrict__ W, int ldw,
    const float* __restrict__ bias,
    float* __restrict__ C, int ldc, int K) {
  int lane = threadIdx.x & 31;
  int wave = threadIdx.x >> 5;
  int row0 = blockIdx.y * 64 + (wave & 3) * 16;
  int col0 = blockIdx.x * 128 + (wave >> 2) * 64;

  v8f acc[4] = {};
  v16bf a0 = load_a_frag(A, lda, row0, 0, lane);
  v16bf b0[4];
#pragma unroll
  for (int i = 0; i < 4; ++i) b0[i] = load_b_frag(W, ldw, col0 + i * 16, 0, lane);
  v16bf a1 = load_a_frag(A, lda, row0, 32, lane);
  v16bf b1[4];
#pragma unroll
  for (int i = 0; i < 4; ++i) b1[i] = load_b_frag(W, ldw, col0 + i * 16, 32, lane);

  for (int k = 64; k < K; k += 64) {
#pragma unroll
    for (int i = 0; i < 4; ++i) acc[i] = wmma_bf16(a0, b0[i], acc[i]);
    a0 = load_a_frag(A, lda, row0, k, lane);
#pragma unroll
    for (int i = 0; i < 4; ++i) b0[i] = load_b_frag(W, ldw, col0 + i * 16, k, lane);
#pragma unroll
    for (int i = 0; i < 4; ++i) acc[i] = wmma_bf16(a1, b1[i], acc[i]);
    a1 = load_a_frag(A, lda, row0, k + 32, lane);
#pragma unroll
    for (int i = 0; i < 4; ++i) b1[i] = load_b_frag(W, ldw, col0 + i * 16, k + 32, lane);
  }
#pragma unroll
  for (int i = 0; i < 4; ++i) acc[i] = wmma_bf16(a0, b0[i], acc[i]);
#pragma unroll
  for (int i = 0; i < 4; ++i) acc[i] = wmma_bf16(a1, b1[i], acc[i]);

  int mr = (lane >> 4) << 3;   // 0 or 8
  int nn = lane & 15;
#pragma unroll
  for (int i = 0; i < 4; ++i) {
    int col = col0 + i * 16 + nn;
    float bv = bias ? bias[col] : 0.0f;
#pragma unroll
    for (int r = 0; r < 8; ++r)
      C[(size_t)(row0 + mr + r) * ldc + col] = acc[i][r] + bv;
  }
}

// ---- Fused recurrent LSTM step ---------------------------------------------
// gates[16 x 1024] = (pre?) + (pre2?) + A0@W0^T + (A1@W1^T?) + (bias?)
// then c' = sig(f)*c + sig(i)*tanh(g); h = sig(o)*tanh(c').
// grid.x = B/16, block = 256 (8 waves; wave w owns cols [w*128, w*128+128)).
// B fragments batched in two groups of 4 so the second group's loads hide
// under the first group's WMMAs.
__global__ __launch_bounds__(256) void lstm_step_kernel(
    const float* __restrict__ pre,    // [B][1024] or null
    const float* __restrict__ pre2,   // [B][1024] or null
    const __bf16* __restrict__ A0, int K0, const __bf16* __restrict__ W0, int ldw0,
    const __bf16* __restrict__ A1, int K1, const __bf16* __restrict__ W1, int ldw1,
    const float* __restrict__ bias,   // [1024] or null
    float* __restrict__ c,            // [B][H] in/out
    __bf16* __restrict__ hOut,        // [B][H] out (bf16, next-step A operand)
    __bf16* __restrict__ hsDst, int hsStride) {  // optional concat dest
  __shared__ float gates[16][G4H];

  int lane = threadIdx.x & 31;
  int wave = threadIdx.x >> 5;
  int row0 = blockIdx.x * 16;
  int col0 = wave * 128;

  v8f acc[8] = {};
  for (int k = 0; k < K0; k += 32) {
    v16bf a = load_a_frag(A0, K0, row0, k, lane);
    v16bf b0[4], b1[4];
#pragma unroll
    for (int i = 0; i < 4; ++i) b0[i] = load_b_frag(W0, ldw0, col0 + i * 16, k, lane);
#pragma unroll
    for (int i = 0; i < 4; ++i) b1[i] = load_b_frag(W0, ldw0, col0 + 64 + i * 16, k, lane);
#pragma unroll
    for (int i = 0; i < 4; ++i) acc[i] = wmma_bf16(a, b0[i], acc[i]);
#pragma unroll
    for (int i = 0; i < 4; ++i) acc[4 + i] = wmma_bf16(a, b1[i], acc[4 + i]);
  }
  if (A1) {
    for (int k = 0; k < K1; k += 32) {
      v16bf a = load_a_frag(A1, K1, row0, k, lane);
      v16bf b0[4], b1[4];
#pragma unroll
      for (int i = 0; i < 4; ++i) b0[i] = load_b_frag(W1, ldw1, col0 + i * 16, k, lane);
#pragma unroll
      for (int i = 0; i < 4; ++i) b1[i] = load_b_frag(W1, ldw1, col0 + 64 + i * 16, k, lane);
#pragma unroll
      for (int i = 0; i < 4; ++i) acc[i] = wmma_bf16(a, b0[i], acc[i]);
#pragma unroll
      for (int i = 0; i < 4; ++i) acc[4 + i] = wmma_bf16(a, b1[i], acc[4 + i]);
    }
  }
  int mr = (lane >> 4) << 3;
  int nn = lane & 15;
#pragma unroll
  for (int i = 0; i < 8; ++i)
#pragma unroll
    for (int r = 0; r < 8; ++r)
      gates[mr + r][col0 + i * 16 + nn] = acc[i][r];
  __syncthreads();

  // gate nonlinearity: 16 rows x 256 cols, 256 threads x 16 iters
#pragma unroll 4
  for (int it = 0; it < 16; ++it) {
    int idx = it * 256 + threadIdx.x;
    int r = idx >> 8;
    int j = idx & 255;
    int b = row0 + r;
    float gi = gates[r][j];
    float gf = gates[r][j + 256];
    float gg = gates[r][j + 512];
    float go = gates[r][j + 768];
    size_t prow = (size_t)b * G4H;
    if (pre)  { gi += pre[prow + j];  gf += pre[prow + j + 256];
                gg += pre[prow + j + 512];  go += pre[prow + j + 768]; }
    if (pre2) { gi += pre2[prow + j]; gf += pre2[prow + j + 256];
                gg += pre2[prow + j + 512]; go += pre2[prow + j + 768]; }
    if (bias) { gi += bias[j]; gf += bias[j + 256];
                gg += bias[j + 512]; go += bias[j + 768]; }
    size_t hidx = (size_t)b * HH + j;
    float cn = sigf(gf) * c[hidx] + sigf(gi) * tanhf(gg);
    float hn = sigf(go) * tanhf(cn);
    c[hidx] = cn;
    hOut[hidx] = (__bf16)hn;
    if (hsDst) hsDst[(size_t)b * hsStride + j] = (__bf16)hn;
  }
}

// ---- Output projection per decoder step (WMMA, 1 block) --------------------
// out[t] = h2[128,256] @ WoH[48,256]^T + outCtx[128,48]; 8 waves = 8 M-tiles.
__global__ __launch_bounds__(256) void out_step_kernel(
    const __bf16* __restrict__ h, const __bf16* __restrict__ WoH,
    const float* __restrict__ outCtx, float* __restrict__ out, int t) {
  int lane = threadIdx.x & 31;
  int wave = threadIdx.x >> 5;
  int row0 = wave * 16;
  v8f acc[3] = {};
  for (int k = 0; k < HH; k += 32) {
    v16bf a = load_a_frag(h, HH, row0, k, lane);
    v16bf b[3];
#pragma unroll
    for (int i = 0; i < 3; ++i) b[i] = load_b_frag(WoH, HH, i * 16, k, lane);
#pragma unroll
    for (int i = 0; i < 3; ++i) acc[i] = wmma_bf16(a, b[i], acc[i]);
  }
  int mr = (lane >> 4) << 3;
  int nn = lane & 15;
#pragma unroll
  for (int i = 0; i < 3; ++i) {
#pragma unroll
    for (int r = 0; r < 8; ++r) {
      int b = row0 + mr + r;
      int v = i * 16 + nn;
      out[((size_t)b * TT + t) * VV + v] = acc[i][r] + outCtx[b * VV + v];
    }
  }
}

// ---- Prep / elementwise kernels --------------------------------------------

// dst[n][k<Kdst] = k<Ksrc ? bf16(src[n*ldsrc + coff + k]) : 0
__global__ void cast_pad_kernel(const float* __restrict__ src, int ldsrc, int coff,
                                int Ksrc, __bf16* __restrict__ dst, int Kdst, int total) {
  int idx = blockIdx.x * 256 + threadIdx.x;
  if (idx >= total) return;
  int n = idx / Kdst, k = idx - n * Kdst;
  float v = (k < Ksrc) ? src[(size_t)n * ldsrc + coff + k] : 0.0f;
  dst[idx] = (__bf16)v;
}

// encoder_input [B,S,48] -> time-major padded bf16 [(s*B+b)][64]
__global__ void enc_in_cast_kernel(const float* __restrict__ src, __bf16* __restrict__ dst) {
  int idx = blockIdx.x * 256 + threadIdx.x;   // S*B*64
  int v = idx & 63; int r = idx >> 6; int b = r & 127; int s = r >> 7;
  float val = (v < VV) ? src[((size_t)b * SS + s) * VV + v] : 0.0f;
  dst[idx] = (__bf16)val;
}

// teacher-forced decoder inputs: din[t] = t==0 ? 0 : emb[target[b,t-1]]
__global__ void din_build_kernel(const int* __restrict__ tgt, const float* __restrict__ emb,
                                 __bf16* __restrict__ dst) {
  int idx = blockIdx.x * 256 + threadIdx.x;   // T*B*64
  int v = idx & 63; int r = idx >> 6; int b = r & 127; int t = r >> 7;
  float val = 0.0f;
  if (t > 0 && v < VV) {
    int tok = tgt[(size_t)b * TT + (t - 1)];
    val = emb[(size_t)tok * VV + v];
  }
  dst[idx] = (__bf16)val;
}

// scores[b][s] = enc_out[s*B+b] . Wa_e   (hq-term & ba are constant over s ->
// drop: softmax-invariant)
__global__ void enc_scores_kernel(const __bf16* __restrict__ xs,
                                  const float* __restrict__ Wa,
                                  float* __restrict__ scores) {
  int idx = blockIdx.x * 256 + threadIdx.x;   // B*S
  int b = idx >> 9, s = idx & 511;
  const __bf16* row = xs + ((size_t)s * BB + b) * (2 * HH);
  float acc = 0.0f;
  for (int k = 0; k < 2 * HH; ++k) acc += (float)row[k] * Wa[HH + k];
  scores[(size_t)b * SS + s] = acc;
}

// per-b softmax over S + weighted sum context (2H) -> f32 + bf16
__global__ __launch_bounds__(256) void attn_context_kernel(
    const float* __restrict__ scores, const __bf16* __restrict__ xs,
    float* __restrict__ ctx, __bf16* __restrict__ ctxBf) {
  __shared__ float red[256];
  __shared__ float w[SS];
  int b = blockIdx.x, tid = threadIdx.x;
  const float* sc = scores + (size_t)b * SS;
  float m = -1e30f;
  for (int s = tid; s < SS; s += 256) m = fmaxf(m, sc[s]);
  red[tid] = m; __syncthreads();
  for (int o = 128; o > 0; o >>= 1) { if (tid < o) red[tid] = fmaxf(red[tid], red[tid + o]); __syncthreads(); }
  m = red[0]; __syncthreads();
  float ssum = 0.0f;
  for (int s = tid; s < SS; s += 256) { float e = __expf(sc[s] - m); w[s] = e; ssum += e; }
  red[tid] = ssum; __syncthreads();
  for (int o = 128; o > 0; o >>= 1) { if (tid < o) red[tid] += red[tid + o]; __syncthreads(); }
  float inv = 1.0f / red[0];
  __syncthreads();
  float a0 = 0.0f, a1 = 0.0f;
  for (int s = 0; s < SS; ++s) {
    float ww = w[s];
    const __bf16* r = xs + ((size_t)s * BB + b) * (2 * HH);
    a0 += ww * (float)r[tid];
    a1 += ww * (float)r[tid + 256];
  }
  a0 *= inv; a1 *= inv;
  ctx[(size_t)b * 512 + tid]        = a0;
  ctx[(size_t)b * 512 + tid + 256]  = a1;
  ctxBf[(size_t)b * 512 + tid]       = (__bf16)a0;
  ctxBf[(size_t)b * 512 + tid + 256] = (__bf16)a1;
}

// outCtx[b][v] = ctx[b] . Wo[v, 256:768] + bo[v]   (constant over t)
__global__ void out_ctx_kernel(const float* __restrict__ ctx, const float* __restrict__ Wo,
                               const float* __restrict__ bo, float* __restrict__ outCtx) {
  int b = blockIdx.x, v = threadIdx.x;
  if (v >= VV) return;
  float acc = bo[v];
  const float* wrow = Wo + (size_t)v * (3 * HH) + HH;
  const float* crow = ctx + (size_t)b * 512;
  for (int k = 0; k < 512; ++k) acc += crow[k] * wrow[k];
  outCtx[b * VV + v] = acc;
}

__global__ void zero_kernel(unsigned int* __restrict__ p, int nwords) {
  int i = blockIdx.x * 256 + threadIdx.x;
  if (i < nwords) p[i] = 0u;
}

// ---------------------------------------------------------------------------
extern "C" void kernel_launch(void* const* d_in, const int* in_sizes, int n_in,
                              void* d_out, int out_size, void* d_ws, size_t ws_size,
                              hipStream_t stream) {
  (void)in_sizes; (void)n_in; (void)out_size; (void)ws_size;
  const float* encIn = (const float*)d_in[0];
  const int*   tgt   = (const int*)d_in[1];
  const float *eWih[3][2], *eWhh[3][2], *eB[3][2];
  int pi = 2;
  for (int l = 0; l < 3; ++l)
    for (int d = 0; d < 2; ++d) {
      eWih[l][d] = (const float*)d_in[pi++];
      eWhh[l][d] = (const float*)d_in[pi++];
      eB[l][d]   = (const float*)d_in[pi++];
    }
  const float *dWih[3], *dWhh[3], *dB[3];
  for (int l = 0; l < 3; ++l) {
    dWih[l] = (const float*)d_in[pi++];
    dWhh[l] = (const float*)d_in[pi++];
    dB[l]   = (const float*)d_in[pi++];
  }
  const float* Wa  = (const float*)d_in[pi++];
  pi++;                                        // ba: softmax-invariant, unused
  const float* Wo  = (const float*)d_in[pi++];
  const float* bo  = (const float*)d_in[pi++];
  const float* emb = (const float*)d_in[pi++];
  float* out = (float*)d_out;

  // ---- workspace carve (~1.0 GB) ----
  char* wsb = (char*)d_ws;
  size_t off = 0;
  auto carve = [&](size_t bytes) -> char* {
    char* p = wsb + off;
    off += (bytes + 255) & ~(size_t)255;
    return p;
  };
  __bf16* eWihBf[3][2]; __bf16* eWhhBf[3][2];
  for (int l = 0; l < 3; ++l)
    for (int d = 0; d < 2; ++d) {
      eWihBf[l][d] = (__bf16*)carve((size_t)G4H * (l == 0 ? 64 : 512) * 2);
      eWhhBf[l][d] = (__bf16*)carve((size_t)G4H * HH * 2);
    }
  __bf16* dWih0EmbBf = (__bf16*)carve((size_t)G4H * 64 * 2);
  __bf16* dWih0CtxBf = (__bf16*)carve((size_t)G4H * 512 * 2);
  __bf16* dWihBf[3];  dWihBf[0] = nullptr;
  __bf16* dWhhBf[3];
  for (int l = 1; l < 3; ++l) dWihBf[l] = (__bf16*)carve((size_t)G4H * HH * 2);
  for (int l = 0; l < 3; ++l) dWhhBf[l] = (__bf16*)carve((size_t)G4H * HH * 2);
  __bf16* WoHBf   = (__bf16*)carve((size_t)VV * HH * 2);
  __bf16* encInBf = (__bf16*)carve((size_t)SS * BB * 64 * 2);
  __bf16* dinBf   = (__bf16*)carve((size_t)TT * BB * 64 * 2);
  __bf16* xsA     = (__bf16*)carve((size_t)SS * BB * 512 * 2);
  __bf16* xsB     = (__bf16*)carve((size_t)SS * BB * 512 * 2);
  float*  PreF    = (float*)carve((size_t)SS * BB * G4H * 4);
  float*  PreB    = (float*)carve((size_t)SS * BB * G4H * 4);
  float*  PreEmb  = (float*)carve((size_t)TT * BB * G4H * 4);
  float*  PreCtx  = (float*)carve((size_t)BB * G4H * 4);
  __bf16* encHf = (__bf16*)carve((size_t)BB * HH * 2);
  __bf16* encHb = (__bf16*)carve((size_t)BB * HH * 2);
  float*  encCf = (float*)carve((size_t)BB * HH * 4);
  float*  encCb = (float*)carve((size_t)BB * HH * 4);
  __bf16* decH[3]; float* decC[3];
  for (int l = 0; l < 3; ++l) {
    decH[l] = (__bf16*)carve((size_t)BB * HH * 2);
    decC[l] = (float*)carve((size_t)BB * HH * 4);
  }
  float*  scores = (float*)carve((size_t)BB * SS * 4);
  float*  ctx    = (float*)carve((size_t)BB * 512 * 4);
  __bf16* ctxBf  = (__bf16*)carve((size_t)BB * 512 * 2);
  float*  outCtx = (float*)carve((size_t)BB * VV * 4);

  auto castpad = [&](const float* src, int ldsrc, int coff, int Ksrc,
                     __bf16* dst, int Kdst, int N) {
    int total = N * Kdst;
    cast_pad_kernel<<<(total + 255) / 256, 256, 0, stream>>>(src, ldsrc, coff, Ksrc,
                                                             dst, Kdst, total);
  };

  // ---- weight prep (f32 -> bf16, pad K=48 -> 64) ----
  for (int d = 0; d < 2; ++d) {
    castpad(eWih[0][d], VV, 0, VV, eWihBf[0][d], 64, G4H);
    castpad(eWih[1][d], 512, 0, 512, eWihBf[1][d], 512, G4H);
    castpad(eWih[2][d], 512, 0, 512, eWihBf[2][d], 512, G4H);
    for (int l = 0; l < 3; ++l)
      castpad(eWhh[l][d], HH, 0, HH, eWhhBf[l][d], HH, G4H);
  }
  castpad(dWih[0], VV + 512, 0, VV, dWih0EmbBf, 64, G4H);
  castpad(dWih[0], VV + 512, VV, 512, dWih0CtxBf, 512, G4H);
  castpad(dWih[1], HH, 0, HH, dWihBf[1], HH, G4H);
  castpad(dWih[2], HH, 0, HH, dWihBf[2], HH, G4H);
  for (int l = 0; l < 3; ++l) castpad(dWhh[l], HH, 0, HH, dWhhBf[l], HH, G4H);
  castpad(Wo, 3 * HH, 0, HH, WoHBf, HH, VV);

  // ---- inputs ----
  enc_in_cast_kernel<<<(SS * BB * 64) / 256, 256, 0, stream>>>(encIn, encInBf);
  din_build_kernel<<<(TT * BB * 64) / 256, 256, 0, stream>>>(tgt, emb, dinBf);

  auto zero = [&](void* p, size_t bytes) {
    int nw = (int)(bytes / 4);
    zero_kernel<<<(nw + 255) / 256, 256, 0, stream>>>((unsigned int*)p, nw);
  };

  const dim3 blk(256);
  const dim3 gemmBig(8, (SS * BB) / 64);   // N=1024/128, M=65536/64

  // ---- encoder ----
  const __bf16* X = encInBf;
  int Kin = 64;
  __bf16* xsOut = xsA;
  for (int l = 0; l < 3; ++l) {
    gemm_bias_kernel<<<gemmBig, blk, 0, stream>>>(X, Kin, eWihBf[l][0], Kin,
                                                  eB[l][0], PreF, G4H, Kin);
    gemm_bias_kernel<<<gemmBig, blk, 0, stream>>>(X, Kin, eWihBf[l][1], Kin,
                                                  eB[l][1], PreB, G4H, Kin);
    zero(encHf, (size_t)BB * HH * 2); zero(encHb, (size_t)BB * HH * 2);
    zero(encCf, (size_t)BB * HH * 4); zero(encCb, (size_t)BB * HH * 4);
    for (int s = 0; s < SS; ++s) {
      int tf = s, tb = SS - 1 - s;
      lstm_step_kernel<<<BB / 16, blk, 0, stream>>>(
          PreF + (size_t)tf * BB * G4H, nullptr,
          encHf, HH, eWhhBf[l][0], HH, nullptr, 0, nullptr, 0, nullptr,
          encCf, encHf, xsOut + (size_t)tf * BB * 512, 512);
      lstm_step_kernel<<<BB / 16, blk, 0, stream>>>(
          PreB + (size_t)tb * BB * G4H, nullptr,
          encHb, HH, eWhhBf[l][1], HH, nullptr, 0, nullptr, 0, nullptr,
          encCb, encHb, xsOut + (size_t)tb * BB * 512 + HH, 512);
    }
    // decoder init states = last L of the 2L (layer,dir) finals:
    // [hTb1, hTf2, hTb2]
    if (l == 1) {
      hipMemcpyAsync(decH[0], encHb, (size_t)BB * HH * 2, hipMemcpyDeviceToDevice, stream);
      hipMemcpyAsync(decC[0], encCb, (size_t)BB * HH * 4, hipMemcpyDeviceToDevice, stream);
    }
    if (l == 2) {
      hipMemcpyAsync(decH[1], encHf, (size_t)BB * HH * 2, hipMemcpyDeviceToDevice, stream);
      hipMemcpyAsync(decC[1], encCf, (size_t)BB * HH * 4, hipMemcpyDeviceToDevice, stream);
      hipMemcpyAsync(decH[2], encHb, (size_t)BB * HH * 2, hipMemcpyDeviceToDevice, stream);
      hipMemcpyAsync(decC[2], encCb, (size_t)BB * HH * 4, hipMemcpyDeviceToDevice, stream);
    }
    X = xsOut;
    Kin = 512;
    xsOut = (l == 0) ? xsB : xsA;
  }
  const __bf16* encOutBf = X;   // layer-2 output lives in xsA

  // ---- attention (step-invariant -> computed once) ----
  enc_scores_kernel<<<(BB * SS) / 256, 256, 0, stream>>>(encOutBf, Wa, scores);
  attn_context_kernel<<<BB, blk, 0, stream>>>(scores, encOutBf, ctx, ctxBf);
  gemm_bias_kernel<<<dim3(8, BB / 64), blk, 0, stream>>>(ctxBf, 512, dWih0CtxBf, 512,
                                                         nullptr, PreCtx, G4H, 512);
  gemm_bias_kernel<<<gemmBig, blk, 0, stream>>>(dinBf, 64, dWih0EmbBf, 64,
                                                dB[0], PreEmb, G4H, 64);
  out_ctx_kernel<<<BB, 64, 0, stream>>>(ctx, Wo, bo, outCtx);

  // ---- decoder scan ----
  for (int t = 0; t < TT; ++t) {
    lstm_step_kernel<<<BB / 16, blk, 0, stream>>>(
        PreEmb + (size_t)t * BB * G4H, PreCtx,
        decH[0], HH, dWhhBf[0], HH, nullptr, 0, nullptr, 0, nullptr,
        decC[0], decH[0], nullptr, 0);
    lstm_step_kernel<<<BB / 16, blk, 0, stream>>>(
        nullptr, nullptr,
        decH[0], HH, dWihBf[1], HH, decH[1], HH, dWhhBf[1], HH, dB[1],
        decC[1], decH[1], nullptr, 0);
    lstm_step_kernel<<<BB / 16, blk, 0, stream>>>(
        nullptr, nullptr,
        decH[1], HH, dWihBf[2], HH, decH[2], HH, dWhhBf[2], HH, dB[2],
        decC[2], decH[2], nullptr, 0);
    out_step_kernel<<<1, blk, 0, stream>>>(decH[2], WoHBf, outCtx, out, t);
  }
}
